// ChessLanguageModel_17343077941444
// MI455X (gfx1250) — matrix operand order repaired and checked by reading, not compiled, not run on hardware
//
#include <hip/hip_runtime.h>
#include <hip/hip_bf16.h>
#include <math.h>

// ---------------------------------------------------------------------------
// Model constants (match reference)
// ---------------------------------------------------------------------------
#define N_CTX    2048
#define D_MODEL  1024
#define D_K      64
#define N_HEADS  16
#define D_HID    4096
#define N_LAYERS 4

// GEMM tiling: 256 threads = 8 waves (wave32), arranged 2 (M) x 4 (N).
// Each wave owns a 32x64 output tile = 2x4 WMMA 16x16 tiles, K-step 32.
// A-fragments are reused across 4 B-fragments -> 1.5 LDS B128 loads per WMMA.
#define BM 64
#define BN 256
#define BK 32
#define GEMM_THREADS 256

typedef unsigned short u16;
typedef __attribute__((ext_vector_type(16))) __bf16 v16bf;
typedef __attribute__((ext_vector_type(8)))  float  v8f;

union FragAB {
    v16bf v;
    uint4 q[2];
};

// fp32 -> bf16 (round to nearest even) as raw bits.
__device__ __forceinline__ u16 f2bf(float f) {
    unsigned u = __float_as_uint(f);
    u += 0x7fffu + ((u >> 16) & 1u);
    return (u16)(u >> 16);
}

// ---------------------------------------------------------------------------
// CDNA5 async global->LDS copy (ASYNCcnt) + wait. Per-lane 16-byte transfer:
//   each lane copies 16 B from its global address to its LDS address     (GLOBAL_LOAD_ASYNC_TO_LDS_B128)
// ---------------------------------------------------------------------------
__device__ __forceinline__ void async_copy_b128(unsigned lds_off, const void* g) {
    asm volatile("global_load_async_to_lds_b128 %0, %1, off"
                 :: "v"(lds_off), "v"((unsigned long long)(uintptr_t)g)
                 : "memory");
}
__device__ __forceinline__ void async_wait0() {
    asm volatile("s_wait_asynccnt 0x0" ::: "memory");
}

// ---------------------------------------------------------------------------
// Tiled bf16 WMMA GEMM, all operands already bf16 in global memory.
//   A   : [M, K] row-major bf16 (lda in elements)
//   Bcm : column-major bf16: column n is the K-contiguous run Bcm[n*ldb + k]
//   C   : optional fp32 out;  Cb: optional bf16 out
//         cb_mode 1: Cb[m*ldcb + n]   (row-major)
//         cb_mode 2: Cb[n*ldcb + m]   (transposed)
//   val = act( outScale * acc + bias[n] )
// Double-buffered LDS, async-copy staging, causal block skip for QK^T.
// ---------------------------------------------------------------------------
__global__ __launch_bounds__(GEMM_THREADS)
void k_gemm_bf16(const u16* __restrict__ A, int lda,
                 const u16* __restrict__ Bcm, int ldb,
                 float* __restrict__ C, int ldc,
                 u16* __restrict__ Cb, int ldcb, int cb_mode,
                 const float* __restrict__ bias,
                 int K, int ncols, float outScale,
                 int act_gelu, int causal)
{
    const int m0 = blockIdx.y * BM;
    const int n0 = blockIdx.x * BN;
    if (causal && n0 > m0 + (BM - 1)) return;   // fully-masked score block

    __shared__ u16 Asm[2][BM][BK];   // [m][k] bf16 bits            (2 x 4KB)
    __shared__ u16 Bsm[2][BN][BK];   // [n][k] bf16 bits K-contig   (2 x 16KB)

    const int tid   = threadIdx.x;
    const int lane  = tid & 31;
    const int wave  = tid >> 5;
    const int wm    = wave >> 2;     // 0..1 (row of 32)
    const int wn    = wave & 3;      // 0..3 (col of 64)
    const int lan16 = lane & 15;
    const int khalf = lane >> 4;     // 0 or 1

    // staging coordinates (contiguous runs per thread)
    const int a_row = tid >> 2;          // 0..63
    const int a_col = (tid & 3) * 8;     // 0,8,16,24
    const int b_n   = tid;               // 0..255: one B column per thread
    const bool b_ok = (n0 + b_n) < ncols;

    v8f acc[2][4] = {};

    const int nt = K / BK;

    // ---- zero OOB B columns once (they stay zero for the whole GEMM) ----
    if (!b_ok) {
        uint4 z = {};
        #pragma unroll
        for (int buf = 0; buf < 2; ++buf)
            #pragma unroll
            for (int j = 0; j < 4; ++j)
                *(uint4*)&Bsm[buf][b_n][j * 8] = z;
    }

    // ---- prologue: stage tile 0 into buffer 0 ----
    {
        async_copy_b128((unsigned)(uintptr_t)&Asm[0][a_row][a_col],
                        A + (size_t)(m0 + a_row) * lda + a_col);
        if (b_ok) {
            unsigned l0 = (unsigned)(uintptr_t)&Bsm[0][b_n][0];
            const u16* g = Bcm + (size_t)(n0 + b_n) * ldb;
            #pragma unroll
            for (int j = 0; j < 4; ++j)
                async_copy_b128(l0 + j * 16, g + j * 8);
        }
    }
    async_wait0();
    __syncthreads();

    for (int i = 0; i < nt; ++i) {
        const int cur = i & 1;

        // ---- stage tile i+1 into the other buffer (read-complete since the
        //      barrier that ended iteration i-1) ----
        if (i + 1 < nt) {
            const int nb = cur ^ 1;
            const int k1 = (i + 1) * BK;
            async_copy_b128((unsigned)(uintptr_t)&Asm[nb][a_row][a_col],
                            A + (size_t)(m0 + a_row) * lda + k1 + a_col);
            if (b_ok) {
                unsigned l0 = (unsigned)(uintptr_t)&Bsm[nb][b_n][0];
                const u16* g = Bcm + (size_t)(n0 + b_n) * ldb + k1;
                #pragma unroll
                for (int j = 0; j < 4; ++j)
                    async_copy_b128(l0 + j * 16, g + j * 8);
            }
        }

        // ---- fragments per CDNA5 16-bit WMMA VGPR layouts ----
        FragAB afrag[2], bfrag[4];
        #pragma unroll
        for (int t = 0; t < 2; ++t) {
            // A 16x32: lanes 0-15 K 0..7/16..23, lanes 16-31 K 8..15/24..31
            int m = wm * 32 + t * 16 + lan16;
            afrag[t].q[0] = *(const uint4*)&Asm[cur][m][khalf * 8];
            afrag[t].q[1] = *(const uint4*)&Asm[cur][m][16 + khalf * 8];
        }
        #pragma unroll
        for (int t = 0; t < 4; ++t) {
            // B 32x16: lanes 0-15 K 0..7/16..23, lanes 16-31 K 8..15/24..31
            int n = wn * 64 + t * 16 + lan16;
            bfrag[t].q[0] = *(const uint4*)&Bsm[cur][n][khalf * 8];
            bfrag[t].q[1] = *(const uint4*)&Bsm[cur][n][16 + khalf * 8];
        }
        #pragma unroll
        for (int tm = 0; tm < 2; ++tm)
            #pragma unroll
            for (int tn = 0; tn < 4; ++tn)
                acc[tm][tn] = __builtin_amdgcn_wmma_f32_16x16x32_bf16(
                    false, afrag[tm].v, false, bfrag[tn].v,
                    (short)0, acc[tm][tn], false, false);

        async_wait0();      // tile i+1 landed
        __syncthreads();    // everyone done reading buffer `cur`
    }

    // ---- epilogue: C/D layout: VGPR r -> M = r + 8*khalf, N = lane&15 ----
    #pragma unroll
    for (int tm = 0; tm < 2; ++tm) {
        #pragma unroll
        for (int tn = 0; tn < 4; ++tn) {
            int n = n0 + wn * 64 + tn * 16 + lan16;
            if (n < ncols) {
                int mbase = m0 + wm * 32 + tm * 16 + khalf * 8;
                #pragma unroll
                for (int r = 0; r < 8; ++r) {
                    float v = acc[tm][tn][r] * outScale;
                    if (bias) v += bias[n];
                    if (act_gelu) v = 0.5f * v * (1.0f + erff(v * 0.70710678118654752f));
                    int m = mbase + r;
                    if (C) C[(size_t)m * ldc + n] = v;
                    if (Cb) {
                        if (cb_mode == 1) Cb[(size_t)m * ldcb + n] = f2bf(v);
                        else              Cb[(size_t)n * ldcb + m] = f2bf(v);
                    }
                }
            }
        }
    }
}

// ---------------------------------------------------------------------------
// Transpose-convert: W[K][Ncol] fp32 row-major -> Wt[Ncol][K] bf16
// ---------------------------------------------------------------------------
__global__ __launch_bounds__(256)
void k_convT(const float* __restrict__ W, u16* __restrict__ Wt, int K, int Ncol)
{
    __shared__ float tile[32][33];
    int k0 = blockIdx.y * 32, n0 = blockIdx.x * 32;
    int tx = threadIdx.x & 31, ty = threadIdx.x >> 5;   // 32 x 8
    for (int i = ty; i < 32; i += 8) {
        int k = k0 + i, n = n0 + tx;
        tile[i][tx] = (k < K && n < Ncol) ? W[(size_t)k * Ncol + n] : 0.0f;
    }
    __syncthreads();
    for (int i = ty; i < 32; i += 8) {
        int n = n0 + i, k = k0 + tx;
        if (n < Ncol && k < K) Wt[(size_t)n * K + k] = f2bf(tile[tx][i]);
    }
}

// ---------------------------------------------------------------------------
// x[i] = emb[seq[i]] + pos[i]  (fp32 + bf16 copies)
// ---------------------------------------------------------------------------
__global__ __launch_bounds__(256)
void k_embed(const int* __restrict__ seq, const float* __restrict__ emb,
             const float* __restrict__ pos, float* __restrict__ x,
             u16* __restrict__ xb)
{
    int row = blockIdx.x, tid = threadIdx.x;
    const float* ep = emb + (size_t)seq[row] * D_MODEL;
    const float* pp = pos + (size_t)row * D_MODEL;
    float* xp = x + (size_t)row * D_MODEL;
    u16*   bp = xb + (size_t)row * D_MODEL;
    for (int i = tid; i < D_MODEL; i += 256) {
        float v = ep[i] + pp[i];
        xp[i] = v;
        bp[i] = f2bf(v);
    }
}

// ---------------------------------------------------------------------------
// Causal softmax over row `blockIdx.x` of S[N_CTX, n]; writes fp32 probs back
// and a bf16 copy (zeros in the masked region) for the P*V GEMM.
// ---------------------------------------------------------------------------
__global__ __launch_bounds__(256)
void k_softmax_causal(float* __restrict__ S, u16* __restrict__ P, int n)
{
    int row = blockIdx.x, tid = threadIdx.x;
    float* sp = S + (size_t)row * n;
    u16*   pp = P + (size_t)row * n;
    __shared__ float red[256];
    int valid = row + 1;

    float mx = -INFINITY;
    for (int j = tid; j < valid; j += 256) mx = fmaxf(mx, sp[j]);
    red[tid] = mx; __syncthreads();
    for (int s = 128; s > 0; s >>= 1) {
        if (tid < s) red[tid] = fmaxf(red[tid], red[tid + s]);
        __syncthreads();
    }
    mx = red[0]; __syncthreads();

    float sum = 0.0f;
    for (int j = tid; j < valid; j += 256) sum += expf(sp[j] - mx);
    red[tid] = sum; __syncthreads();
    for (int s = 128; s > 0; s >>= 1) {
        if (tid < s) red[tid] += red[tid + s];
        __syncthreads();
    }
    float inv = 1.0f / red[0];

    for (int j = tid; j < n; j += 256) {
        float p = (j < valid) ? expf(sp[j] - mx) * inv : 0.0f;
        sp[j] = p;
        pp[j] = f2bf(p);
    }
}

// ---------------------------------------------------------------------------
// h = LayerNorm(h + t) * g + b   (biased variance, eps = 1e-5) + bf16 copy
// ---------------------------------------------------------------------------
__global__ __launch_bounds__(256)
void k_add_ln(float* __restrict__ h, const float* __restrict__ t,
              const float* __restrict__ g, const float* __restrict__ b,
              u16* __restrict__ hb)
{
    int row = blockIdx.x, tid = threadIdx.x;
    float* hp = h + (size_t)row * D_MODEL;
    const float* tp = t + (size_t)row * D_MODEL;
    u16* bp = hb + (size_t)row * D_MODEL;
    __shared__ float rs[256], rss[256];

    float s = 0.0f, ss = 0.0f;
    for (int i = tid; i < D_MODEL; i += 256) {
        float x = hp[i] + tp[i];
        s += x; ss += x * x;
    }
    rs[tid] = s; rss[tid] = ss; __syncthreads();
    for (int k = 128; k > 0; k >>= 1) {
        if (tid < k) { rs[tid] += rs[tid + k]; rss[tid] += rss[tid + k]; }
        __syncthreads();
    }
    float mean = rs[0] * (1.0f / D_MODEL);
    float var  = rss[0] * (1.0f / D_MODEL) - mean * mean;
    float inv  = rsqrtf(var + 1e-5f);
    for (int i = tid; i < D_MODEL; i += 256) {
        float x = hp[i] + tp[i];
        float y = (x - mean) * inv * g[i] + b[i];
        hp[i] = y;
        bp[i] = f2bf(y);
    }
}

// ---------------------------------------------------------------------------
// Cross-entropy heads (loss = nll / log(vocab))
// ---------------------------------------------------------------------------
__global__ __launch_bounds__(256)
void k_ce_seq(const float* __restrict__ logits, const int* __restrict__ tgt,
              float* __restrict__ out)
{
    int row = blockIdx.x, tid = threadIdx.x;   // 256 threads == vocab
    const float* lp = logits + (size_t)row * 256;
    __shared__ float red[256];
    float v = lp[tid];
    red[tid] = v; __syncthreads();
    for (int s = 128; s > 0; s >>= 1) {
        if (tid < s) red[tid] = fmaxf(red[tid], red[tid + s]);
        __syncthreads();
    }
    float mx = red[0]; __syncthreads();
    red[tid] = expf(v - mx); __syncthreads();
    for (int s = 128; s > 0; s >>= 1) {
        if (tid < s) red[tid] += red[tid + s];
        __syncthreads();
    }
    if (tid == 0) {
        float lse = logf(red[0]) + mx;
        out[row] = (lse - lp[tgt[row]]) * (1.0f / logf(256.0f));
    }
}

__global__ __launch_bounds__(64)
void k_ce_visual(const float* __restrict__ logits, const int* __restrict__ tgt,
                 float* __restrict__ out)
{
    int row = blockIdx.x, g = threadIdx.x;     // 64 groups of 13
    const float* lp = logits + (size_t)row * 832 + g * 13;
    float mx = -INFINITY;
    #pragma unroll
    for (int j = 0; j < 13; ++j) mx = fmaxf(mx, lp[j]);
    float sum = 0.0f;
    #pragma unroll
    for (int j = 0; j < 13; ++j) sum += expf(lp[j] - mx);
    int t = tgt[(size_t)row * 64 + g];
    out[(size_t)row * 64 + g] = (logf(sum) + mx - lp[t]) * (1.0f / logf(13.0f));
}

__global__ __launch_bounds__(256)
void k_ce_action(const float* __restrict__ logits, const int* __restrict__ tgt,
                 float* __restrict__ out)
{
    int row = blockIdx.x, g = threadIdx.x;     // 256 groups of 2
    const float* lp = logits + (size_t)row * 512 + g * 2;
    float a = lp[0], b = lp[1];
    float mx = fmaxf(a, b);
    float sum = expf(a - mx) + expf(b - mx);
    int t = tgt[(size_t)row * 256 + g];
    out[(size_t)row * 256 + g] = (logf(sum) + mx - lp[t]) * (1.0f / logf(2.0f));
}

// ---------------------------------------------------------------------------
// Host orchestration
// ---------------------------------------------------------------------------
static inline void launch_gemm(hipStream_t s,
                               const u16* A, int lda,
                               const u16* Bcm, int ldb,
                               float* C, int ldc,
                               u16* Cb, int ldcb, int cb_mode,
                               const float* bias,
                               int K, int ncols, float outScale,
                               int act_gelu, int causal)
{
    dim3 grid((ncols + BN - 1) / BN, N_CTX / BM);
    k_gemm_bf16<<<grid, GEMM_THREADS, 0, s>>>(A, lda, Bcm, ldb, C, ldc,
                                              Cb, ldcb, cb_mode, bias,
                                              K, ncols, outScale,
                                              act_gelu, causal);
}

static inline void launch_convT(hipStream_t s, const float* W, u16* Wt,
                                int K, int Ncol)
{
    dim3 grid((Ncol + 31) / 32, (K + 31) / 32);
    k_convT<<<grid, 256, 0, s>>>(W, Wt, K, Ncol);
}

extern "C" void kernel_launch(void* const* d_in, const int* in_sizes, int n_in,
                              void* d_out, int out_size, void* d_ws, size_t ws_size,
                              hipStream_t stream)
{
    (void)in_sizes; (void)n_in; (void)out_size; (void)ws_size;

    const int*   seq_input     = (const int*)d_in[0];
    const int*   seq_target    = (const int*)d_in[1];
    const int*   visual_target = (const int*)d_in[2];
    const int*   action_target = (const int*)d_in[3];
    const float* emb  = (const float*)d_in[4];
    const float* pos  = (const float*)d_in[5];
    const float* Wq   = (const float*)d_in[6];
    const float* bq   = (const float*)d_in[7];
    const float* Wk   = (const float*)d_in[8];
    const float* bk   = (const float*)d_in[9];
    const float* Wv   = (const float*)d_in[10];
    const float* bv   = (const float*)d_in[11];
    const float* Wo   = (const float*)d_in[12];
    const float* W1   = (const float*)d_in[13];
    const float* b1   = (const float*)d_in[14];
    const float* W2   = (const float*)d_in[15];
    const float* b2   = (const float*)d_in[16];
    const float* ln1g = (const float*)d_in[17];
    const float* ln1b = (const float*)d_in[18];
    const float* ln2g = (const float*)d_in[19];
    const float* ln2b = (const float*)d_in[20];
    const float* Ws   = (const float*)d_in[21];
    const float* bs   = (const float*)d_in[22];
    const float* Wvh  = (const float*)d_in[23];
    const float* bvh  = (const float*)d_in[24];
    const float* Wa   = (const float*)d_in[25];
    const float* ba   = (const float*)d_in[26];

    float* out = (float*)d_out;

    // ---- workspace carve-up (256B aligned regions) ----
    char* wsb = (char*)d_ws;
    size_t off = 0;
    auto allocF = [&](size_t n) {
        float* p = (float*)(wsb + off);
        off = (off + n * 4 + 255) & ~(size_t)255;
        return p;
    };
    auto allocH = [&](size_t n) {
        u16* p = (u16*)(wsb + off);
        off = (off + n * 2 + 255) & ~(size_t)255;
        return p;
    };

    const size_t ND = (size_t)N_CTX * D_MODEL;
    const int HD = N_HEADS * D_K;   // 1024

    float* x      = allocF(ND);                        // residual stream fp32
    float* tmp    = allocF(ND);                        // sub-block output
    float* scores = allocF((size_t)N_CTX * N_CTX);     // per-head scores
    float* logits = allocF((size_t)N_CTX * 832);       // head logits (max 832)

    u16* xb     = allocH(ND);                          // bf16 residual
    u16* qb16   = allocH(ND);                          // Q bf16 row-major
    u16* kb16   = allocH(ND);                          // K bf16 row-major
    u16* vt16   = allocH(ND);                          // V bf16 transposed [HD][N]
    u16* ctx16  = allocH(ND);                          // head contexts bf16
    u16* probs  = allocH((size_t)N_CTX * N_CTX);       // probs bf16
    u16* hid16  = allocH((size_t)N_CTX * D_HID);       // FFN hidden bf16

    const size_t WSZ = (size_t)D_MODEL * HD;           // 1M per layer
    const size_t FSZ = (size_t)D_MODEL * D_HID;        // 4M per layer
    u16* wqt = allocH(N_LAYERS * WSZ);
    u16* wkt = allocH(N_LAYERS * WSZ);
    u16* wvt = allocH(N_LAYERS * WSZ);
    u16* wot = allocH(N_LAYERS * WSZ);
    u16* w1t = allocH(N_LAYERS * FSZ);
    u16* w2t = allocH(N_LAYERS * FSZ);
    u16* wst  = allocH((size_t)256 * D_MODEL);
    u16* wvht = allocH((size_t)832 * D_MODEL);
    u16* wat  = allocH((size_t)512 * D_MODEL);

    // ---- one-time weight transpose-convert fp32 -> column-major bf16 ----
    for (int l = 0; l < N_LAYERS; ++l) {
        launch_convT(stream, Wq + l * WSZ, wqt + l * WSZ, D_MODEL, HD);
        launch_convT(stream, Wk + l * WSZ, wkt + l * WSZ, D_MODEL, HD);
        launch_convT(stream, Wv + l * WSZ, wvt + l * WSZ, D_MODEL, HD);
        launch_convT(stream, Wo + l * WSZ, wot + l * WSZ, HD, D_MODEL);
        launch_convT(stream, W1 + (size_t)l * FSZ, w1t + (size_t)l * FSZ,
                     D_MODEL, D_HID);
        launch_convT(stream, W2 + (size_t)l * FSZ, w2t + (size_t)l * FSZ,
                     D_HID, D_MODEL);
    }
    launch_convT(stream, Ws,  wst,  D_MODEL, 256);
    launch_convT(stream, Wvh, wvht, D_MODEL, 832);
    launch_convT(stream, Wa,  wat,  D_MODEL, 512);

    // ---- embedding + positions ----
    k_embed<<<N_CTX, 256, 0, stream>>>(seq_input, emb, pos, x, xb);

    // ---- transformer layers ----
    for (int l = 0; l < N_LAYERS; ++l) {
        // QKV projections: bf16 outputs only (q/k row-major, v transposed)
        launch_gemm(stream, xb, D_MODEL, wqt + l * WSZ, D_MODEL,
                    nullptr, 0, qb16, HD, 1, bq + (size_t)l * HD,
                    D_MODEL, HD, 1.0f, 0, 0);
        launch_gemm(stream, xb, D_MODEL, wkt + l * WSZ, D_MODEL,
                    nullptr, 0, kb16, HD, 1, bk + (size_t)l * HD,
                    D_MODEL, HD, 1.0f, 0, 0);
        launch_gemm(stream, xb, D_MODEL, wvt + l * WSZ, D_MODEL,
                    nullptr, 0, vt16, N_CTX, 2, bv + (size_t)l * HD,
                    D_MODEL, HD, 1.0f, 0, 0);

        // per-head causal attention
        for (int h = 0; h < N_HEADS; ++h) {
            // S = (Q_h * K_h^T)/sqrt(dk): B column n = key n's K-run in kb16
            launch_gemm(stream, qb16 + h * D_K, HD, kb16 + h * D_K, HD,
                        scores, N_CTX, nullptr, 0, 0, nullptr,
                        D_K, N_CTX, 0.125f, 0, /*causal=*/1);
            k_softmax_causal<<<N_CTX, 256, 0, stream>>>(scores, probs, N_CTX);
            // ctx_h = P * V_h : B column n = vt16 row (h*64+n)
            launch_gemm(stream, probs, N_CTX,
                        vt16 + (size_t)h * D_K * N_CTX, N_CTX,
                        nullptr, 0, ctx16 + h * D_K, HD, 1, nullptr,
                        N_CTX, D_K, 1.0f, 0, 0);
        }

        // output projection + residual + LN1
        launch_gemm(stream, ctx16, HD, wot + l * WSZ, HD,
                    tmp, D_MODEL, nullptr, 0, 0, nullptr,
                    HD, D_MODEL, 1.0f, 0, 0);
        k_add_ln<<<N_CTX, 256, 0, stream>>>(x, tmp,
                                            ln1g + (size_t)l * D_MODEL,
                                            ln1b + (size_t)l * D_MODEL, xb);

        // FFN: hidden = GELU(x W1 + b1) (bf16), tmp = hidden W2 + b2 (fp32)
        launch_gemm(stream, xb, D_MODEL, w1t + (size_t)l * FSZ, D_MODEL,
                    nullptr, 0, hid16, D_HID, 1, b1 + (size_t)l * D_HID,
                    D_MODEL, D_HID, 1.0f, /*gelu=*/1, 0);
        launch_gemm(stream, hid16, D_HID, w2t + (size_t)l * FSZ, D_HID,
                    tmp, D_MODEL, nullptr, 0, 0, b2 + (size_t)l * D_MODEL,
                    D_HID, D_MODEL, 1.0f, 0, 0);
        k_add_ln<<<N_CTX, 256, 0, stream>>>(x, tmp,
                                            ln2g + (size_t)l * D_MODEL,
                                            ln2b + (size_t)l * D_MODEL, xb);
    }

    // ---- heads + losses ----
    launch_gemm(stream, xb, D_MODEL, wst, D_MODEL, logits, 256,
                nullptr, 0, 0, bs, D_MODEL, 256, 1.0f, 0, 0);
    k_ce_seq<<<N_CTX, 256, 0, stream>>>(logits, seq_target, out);

    launch_gemm(stream, xb, D_MODEL, wvht, D_MODEL, logits, 832,
                nullptr, 0, 0, bvh, D_MODEL, 832, 1.0f, 0, 0);
    k_ce_visual<<<N_CTX, 64, 0, stream>>>(logits, visual_target, out + N_CTX);

    launch_gemm(stream, xb, D_MODEL, wat, D_MODEL, logits, 512,
                nullptr, 0, 0, ba, D_MODEL, 512, 1.0f, 0, 0);
    k_ce_action<<<N_CTX, 256, 0, stream>>>(logits, action_target,
                                           out + N_CTX + (size_t)N_CTX * 64);
}
